// BiEncoder_44985487459122
// MI455X (gfx1250) — matrix operand order repaired and checked by reading, not compiled, not run on hardware
//
#include <hip/hip_runtime.h>

#define LSEQ 2048
#define BB   64
#define HH   256
#define EE   256
#define G3   768   // 3*H, gate order [r, z, h]

typedef __attribute__((ext_vector_type(16))) __bf16       v16bf;
typedef __attribute__((ext_vector_type(8)))  float        v8f;
typedef __attribute__((ext_vector_type(4)))  unsigned int u32x4;
typedef __attribute__((ext_vector_type(8)))  unsigned int u32x8;
typedef __attribute__((ext_vector_type(4)))  float        f32x4;

static __device__ __forceinline__ unsigned short f2bf(float f) {
  unsigned int u = __float_as_uint(f);
  u += 0x7FFFu + ((u >> 16) & 1u);          // round-to-nearest-even
  return (unsigned short)(u >> 16);
}

static __device__ __forceinline__ float sigmoid_f(float x) {
  return 1.0f / (1.0f + __expf(-x));
}
static __device__ __forceinline__ float tanh_f(float x) {
  return 1.0f - 2.0f / (__expf(2.0f * x) + 1.0f);
}

// 16 contiguous bf16 (32B, aligned) -> v16bf
static __device__ __forceinline__ v16bf load16bf(const unsigned short* p) {
  u32x4 lo = *(const u32x4*)p;
  u32x4 hi = *(const u32x4*)(p + 8);
  u32x8 c  = __builtin_shufflevector(lo, hi, 0, 1, 2, 3, 4, 5, 6, 7);
  return __builtin_bit_cast(v16bf, c);
}

// A-fragment for V_WMMA_F32_16X16X32_BF16 from row-major [16 x K] tile.
// row = base of this lane's M-row; kh = 0 (lanes 0-15) / 8 (lanes 16-31).
// Elements 0..7  = K kt+kh .. kt+kh+7
// Elements 8..15 = K kt+kh+16 .. kt+kh+23        (ISA 7.12.2 16-bit A layout)
static __device__ __forceinline__ v16bf load_a_frag(const unsigned short* row,
                                                    int kt, int kh) {
  u32x4 lo = *(const u32x4*)(row + kt + kh);
  u32x4 hi = *(const u32x4*)(row + kt + kh + 16);
  u32x8 c  = __builtin_shufflevector(lo, hi, 0, 1, 2, 3, 4, 5, 6, 7);
  return __builtin_bit_cast(v16bf, c);
}

static __device__ __forceinline__ v8f wmma_bf16(v16bf a, v16bf b, v8f c) {
  return __builtin_amdgcn_wmma_f32_16x16x32_bf16(
      false, a, false, b, (short)0, c, false, false);
}

// ---------------- small utility kernels ----------------

__global__ void pack_bf16(const float* __restrict__ src,
                          unsigned short* __restrict__ dst, int n) {
  int i = blockIdx.x * blockDim.x + threadIdx.x;
  if (i < n) dst[i] = f2bf(src[i]);
}

__global__ void zero_f32x4(float* __restrict__ p, long n4) {
  long i = (long)blockIdx.x * blockDim.x + threadIdx.x;
  if (i < n4) {
    f32x4 z = {};
    __builtin_nontemporal_store(z, ((f32x4*)p) + i);
  }
}

__global__ void concat_bias(const float* __restrict__ br0, const float* __restrict__ bz0,
                            const float* __restrict__ bh0, const float* __restrict__ br1,
                            const float* __restrict__ bz1, const float* __restrict__ bh1,
                            float* __restrict__ bcat) {
  int i = blockIdx.x * blockDim.x + threadIdx.x;
  if (i >= 2 * G3) return;
  int d = i / G3, n = i % G3;
  const float* src;
  if (n < HH)          { src = d ? br1 : br0; }
  else if (n < 2 * HH) { src = d ? bz1 : bz0; n -= HH; }
  else                 { src = d ? bh1 : bh0; n -= 2 * HH; }
  bcat[i] = src[n];
}

// ---------------- phase B: x-projection GEMM (streaming, parallel) ----------
// Gx[dir][t][b][n] = emb[tok(dir,t,b)] @ Wpack[dir][n]^T + bias[n]
// M = L*B = 131072 rows, K = 256, N = 768, 16x16x32 bf16 WMMA per wave tile.
__global__ void __launch_bounds__(256, 2)
xproj_wmma(const int* __restrict__ tokens, const int* __restrict__ lengths,
           const float* __restrict__ emb,
           const unsigned short* __restrict__ Wpack,  // [2][768][256] bf16, N-major
           const float* __restrict__ bcat,            // [2][768]
           float* __restrict__ Gx) {                  // [2][L][B][768]
  const int dir  = blockIdx.y;
  const int wave = threadIdx.x >> 5;
  const int lane = threadIdx.x & 31;
  const int mt   = blockIdx.x / 6;                  // 0..8191
  const int nt   = (blockIdx.x % 6) * 8 + wave;     // 0..47

  const int mrow = mt * 16 + (lane & 15);           // global M row (= t*B + b)
  const int kh   = (lane & 16) ? 8 : 0;             // A-frag K sub-offset
  const int bkh  = (lane & 16) ? 8 : 0;             // B-frag K sub-offset
  const int chi  = (lane & 16) ? 8 : 0;             // C row sub-offset
  const int bnk  = lane & 15;                       // C/B column within tile

  const int bcol  = mrow & (BB - 1);
  const int trow0 = mrow >> 6;
  int trow = trow0;
  if (dir) {                                        // rtl: reversed, clipped index
    int r = lengths[bcol] - 1 - trow0;
    trow  = r < 0 ? 0 : (r > LSEQ - 1 ? LSEQ - 1 : r);
  }
  const int    tok  = tokens[trow * BB + bcol];
  const float* arow = emb + (size_t)tok * EE;

  const int ncol = nt * 16 + bnk;
  const unsigned short* brow = Wpack + ((size_t)dir * G3 + ncol) * EE + bkh;

  v8f acc = {};
  #pragma unroll
  for (int kt = 0; kt < EE; kt += 32) {
    v16bf af;
    const float* p = arow + kt + kh;
    #pragma unroll
    for (int i = 0; i < 8; ++i) af[i] = (__bf16)p[i];
    #pragma unroll
    for (int i = 0; i < 8; ++i) af[8 + i] = (__bf16)p[16 + i];
    v16bf bfv = load_a_frag(brow, kt, 0);
    acc = wmma_bf16(af, bfv, acc);
  }

  const float bias = bcat[dir * G3 + ncol];
  float* gout = Gx + (size_t)dir * LSEQ * BB * G3;
  #pragma unroll
  for (int j = 0; j < 8; ++j) {
    int m = mt * 16 + chi + j;
    __builtin_nontemporal_store(acc[j] + bias, gout + (size_t)m * G3 + ncol);
  }
}

// ---------------- phase C: recurrent scan (serial over t) -------------------
// One persistent 1024-thread workgroup (32 waves) per direction.
// LDS: hbf[64][256] bf16 (32KB) + rhbf[64][256] bf16 (32KB) + gates[64][768] f32 (192KB)
__global__ void __launch_bounds__(1024, 1)
gru_scan(const int* __restrict__ lengths,
         const unsigned short* __restrict__ Upack,   // [2][768][256] bf16, N-major
         const float* __restrict__ Gx,               // [2][L][B][768]
         float* __restrict__ out) {                  // [L][B][512]
  extern __shared__ unsigned char smem[];
  unsigned short* hbf  = (unsigned short*)smem;            // h  (bf16 mirror)
  unsigned short* rhbf = hbf + BB * HH;                    // r*h (bf16)
  float* gates = (float*)(smem + (size_t)2 * BB * HH * sizeof(unsigned short));

  const int dir  = blockIdx.x;
  const int tid  = threadIdx.x;
  const int wave = tid >> 5;
  const int lane = tid & 31;
  const int mrow = lane & 15;
  const int kh   = (lane & 16) ? 8 : 0;
  const int bkh  = (lane & 16) ? 8 : 0;
  const int chi  = (lane & 16) ? 8 : 0;
  const int bnk  = lane & 15;

  // elementwise ownership: 16 threads per batch row, 16 h-dims per thread
  const int b   = tid >> 4;
  const int hc0 = (tid & 15) << 4;
  const int len = lengths[b];

  float hreg[16], zreg[16];
  #pragma unroll
  for (int j = 0; j < 16; ++j) { hreg[j] = 0.f; zreg[j] = 0.f; }
  #pragma unroll
  for (int j = 0; j < 16; ++j) hbf[b * HH + hc0 + j] = 0;
  __syncthreads();

  const unsigned short* U      = Upack + (size_t)dir * G3 * HH;
  const float*          gxbase = Gx + (size_t)dir * LSEQ * BB * G3;

  for (int t = 0; t < LSEQ; ++t) {
    // phase 1: r,z GEMM  C[64x512] = h_bf16[64x256] @ U[:,0:512]
    #pragma unroll
    for (int i = 0; i < 4; ++i) {
      const int tt = wave * 4 + i;            // 0..127 tiles
      const int mt = tt & 3;
      const int nt = tt >> 2;                 // 0..31
      const unsigned short* arow = hbf + (mt * 16 + mrow) * HH;
      const unsigned short* brow = U + (size_t)(nt * 16 + bnk) * HH + bkh;
      v8f acc = {};
      #pragma unroll
      for (int kt = 0; kt < HH; kt += 32) {
        v16bf af  = load_a_frag(arow, kt, kh);
        v16bf bfv = load_a_frag(brow, kt, 0);
        acc = wmma_bf16(af, bfv, acc);
      }
      #pragma unroll
      for (int j = 0; j < 8; ++j)
        gates[(mt * 16 + chi + j) * G3 + nt * 16 + bnk] = acc[j];
    }
    __syncthreads();

    // phase 2: r,z elementwise; build (r*h) bf16 for the candidate GEMM
    {
      const float* gx = gxbase + ((size_t)t * BB + b) * G3;
      if (t + 1 < LSEQ) __builtin_prefetch(gx + BB * G3, 0, 0);
      #pragma unroll
      for (int j = 0; j < 16; ++j) {
        const int hd = hc0 + j;
        float r = sigmoid_f(__builtin_nontemporal_load(gx + hd) +
                            gates[b * G3 + hd]);
        float z = sigmoid_f(__builtin_nontemporal_load(gx + HH + hd) +
                            gates[b * G3 + HH + hd]);
        zreg[j] = z;
        rhbf[b * HH + hd] = f2bf(r * hreg[j]);   // reference: (r*h) @ Uh^T
      }
    }
    __syncthreads();

    // phase 3: candidate GEMM  C[64x256] = (r*h)_bf16 @ U[:,512:768]
    #pragma unroll
    for (int i = 0; i < 2; ++i) {
      const int tt = wave * 2 + i;            // 0..63 tiles
      const int mt = tt & 3;
      const int nt = 32 + (tt >> 2);          // 32..47
      const unsigned short* arow = rhbf + (mt * 16 + mrow) * HH;
      const unsigned short* brow = U + (size_t)(nt * 16 + bnk) * HH + bkh;
      v8f acc = {};
      #pragma unroll
      for (int kt = 0; kt < HH; kt += 32) {
        v16bf af  = load_a_frag(arow, kt, kh);
        v16bf bfv = load_a_frag(brow, kt, 0);
        acc = wmma_bf16(af, bfv, acc);
      }
      #pragma unroll
      for (int j = 0; j < 8; ++j)
        gates[(mt * 16 + chi + j) * G3 + nt * 16 + bnk] = acc[j];
    }
    __syncthreads();

    // phase 4: h update + masked output
    {
      const float* gx     = gxbase + ((size_t)t * BB + b) * G3;
      const bool   active = t < len;
      const int    orow   = dir ? (len - 1 - t) : t;   // rtl reversal in-place
      float* op = out + ((size_t)orow * BB + b) * (2 * HH) + dir * HH + hc0;
      #pragma unroll
      for (int j = 0; j < 16; ++j) {
        const int hd = hc0 + j;
        float hp = tanh_f(__builtin_nontemporal_load(gx + 2 * HH + hd) +
                          gates[b * G3 + 2 * HH + hd]);
        float hn = zreg[j] * hp + (1.f - zreg[j]) * hreg[j];
        if (active) hreg[j] = hn;                       // h frozen after seq end
        hbf[b * HH + hd] = f2bf(hreg[j]);
      }
      if (active) {
        #pragma unroll
        for (int j = 0; j < 16; ++j)
          __builtin_nontemporal_store(hreg[j], op + j);
      }
    }
    __syncthreads();
  }
}

// ---------------- host launcher ----------------

extern "C" void kernel_launch(void* const* d_in, const int* in_sizes, int n_in,
                              void* d_out, int out_size, void* d_ws, size_t ws_size,
                              hipStream_t stream) {
  (void)in_sizes; (void)n_in; (void)ws_size;
  const int*   tokens  = (const int*)d_in[0];
  const int*   lengths = (const int*)d_in[1];
  const float* emb     = (const float*)d_in[2];

  // workspace layout
  float* Gx = (float*)d_ws;                                       // 805 MB
  size_t gx_bytes = (size_t)2 * LSEQ * BB * G3 * sizeof(float);
  unsigned short* Wpack = (unsigned short*)((char*)d_ws + gx_bytes);
  unsigned short* Upack = Wpack + (size_t)2 * G3 * EE;
  float* bcat = (float*)(Upack + (size_t)2 * G3 * HH);

  // pack weights (gate order r,z,h), convert f32 -> bf16, N-major rows
  for (int d = 0; d < 2; ++d) {
    int base = 3 + 9 * d;  // Wh,Wz,Wr,Uh,Uz,Ur,bh,bz,br
    const float* Wh = (const float*)d_in[base + 0];
    const float* Wz = (const float*)d_in[base + 1];
    const float* Wr = (const float*)d_in[base + 2];
    const float* Uh = (const float*)d_in[base + 3];
    const float* Uz = (const float*)d_in[base + 4];
    const float* Ur = (const float*)d_in[base + 5];
    const int n = HH * EE;
    dim3 pg((n + 255) / 256);
    unsigned short* wd = Wpack + (size_t)d * G3 * EE;
    unsigned short* ud = Upack + (size_t)d * G3 * HH;
    pack_bf16<<<pg, 256, 0, stream>>>(Wr, wd + 0 * n, n);
    pack_bf16<<<pg, 256, 0, stream>>>(Wz, wd + 1 * n, n);
    pack_bf16<<<pg, 256, 0, stream>>>(Wh, wd + 2 * n, n);
    pack_bf16<<<pg, 256, 0, stream>>>(Ur, ud + 0 * n, n);
    pack_bf16<<<pg, 256, 0, stream>>>(Uz, ud + 1 * n, n);
    pack_bf16<<<pg, 256, 0, stream>>>(Uh, ud + 2 * n, n);
  }
  concat_bias<<<(2 * G3 + 255) / 256, 256, 0, stream>>>(
      (const float*)d_in[11], (const float*)d_in[10], (const float*)d_in[9],
      (const float*)d_in[20], (const float*)d_in[19], (const float*)d_in[18],
      bcat);

  // zero output (masked tail rows must read 0)
  long n4 = (long)out_size / 4;
  zero_f32x4<<<(unsigned)((n4 + 255) / 256), 256, 0, stream>>>((float*)d_out, n4);

  // streaming input-gate GEMM: grid 8192 M-tiles x 6 (each block = 8 N-tiles), 2 dirs
  dim3 xg((LSEQ * BB / 16) * (G3 / 128), 2);
  xproj_wmma<<<xg, 256, 0, stream>>>(tokens, lengths, emb, Wpack, bcat, Gx);

  // persistent recurrent scan: 1 workgroup per direction, 256 KB dynamic LDS
  size_t lds = (size_t)2 * BB * HH * sizeof(unsigned short) +
               (size_t)BB * G3 * sizeof(float);
  gru_scan<<<2, 1024, lds, stream>>>(lengths, Upack, Gx, (float*)d_out);
}
